// Block_15118284882621
// MI455X (gfx1250) — matrix operand order repaired and checked by reading, not compiled, not run on hardware
//
#include <hip/hip_runtime.h>
#include <math.h>

// Problem constants (from reference)
#define B_   4
#define N_   2048
#define C_   1024
#define H_   16
#define HD_  64
#define MT_  40
#define HID_ 4096
#define SCALE_ 0.125f   // 64^-0.5
#define EPS_ 1e-5f

#if defined(__has_builtin)
#if __has_builtin(__builtin_amdgcn_global_load_async_to_lds_b128) && \
    __has_builtin(__builtin_amdgcn_s_wait_asynccnt)
#define USE_ASYNC_LDS 1
#endif
#endif
#ifndef USE_ASYNC_LDS
#define USE_ASYNC_LDS 0
#endif

typedef __attribute__((ext_vector_type(16))) __bf16 bf16x16;
typedef __attribute__((ext_vector_type(8)))  float  f32x8;
typedef __attribute__((ext_vector_type(4)))  int    i32x4;

#if USE_ASYNC_LDS
// The builtin wants int4 pointers qualified with the global (AS1) and
// LDS (AS3) address spaces (diagnostic: "int __vector(4) __device__ *").
typedef __attribute__((address_space(1))) i32x4 gas_i32x4;
typedef __attribute__((address_space(3))) i32x4 las_i32x4;
#endif

// ---------- helpers ----------
__device__ __forceinline__ __bf16 f2b(float f) {
  union { float f; unsigned u; } a; a.f = f;
  unsigned u = a.u;
  unsigned r = (u + 0x7FFFu + ((u >> 16) & 1u)) >> 16;   // RNE
  union { unsigned short s; __bf16 b; } o; o.s = (unsigned short)r;
  return o.b;
}

__device__ __forceinline__ f32x8 zero8() {
  f32x8 z = {0.f,0.f,0.f,0.f,0.f,0.f,0.f,0.f};
  return z;
}

// 16-byte global -> LDS copy. Async (ASYNCcnt-tracked DMA, no VGPR round trip)
// when the toolchain exposes the gfx1250 builtin; plain copy otherwise.
__device__ __forceinline__ void copy16_async(const void* g, void* l) {
#if USE_ASYNC_LDS
  __builtin_amdgcn_global_load_async_to_lds_b128((gas_i32x4*)g, (las_i32x4*)l, 0, 0);
#else
  *(uint4*)l = *(const uint4*)g;
#endif
}

// Wait until at most Nrem async copies outstanding (immediate operand).
template <int Nrem>
__device__ __forceinline__ void async_wait() {
#if USE_ASYNC_LDS
  __builtin_amdgcn_s_wait_asynccnt((unsigned short)Nrem);
#endif
}

// seg: 0 = all rows (r in [0,B*N)), 1 = token rows (n<MT), 2 = image rows (n>=MT)
__device__ __forceinline__ int map_row(int r, int seg, bool& valid, int rows) {
  valid = (r < rows);
  int rr = valid ? r : 0;   // clamp: keeps loads safe + async issue count uniform
  if (seg == 1) { int b = rr / MT_;        int n = rr % MT_;               return b * N_ + n; }
  if (seg == 2) { int b = rr / (N_ - MT_); int n = MT_ + rr % (N_ - MT_);  return b * N_ + n; }
  return rr;
}

// ---------- fp32 -> bf16 weight conversion ----------
__global__ __launch_bounds__(256) void cvt_bf16(const float* __restrict__ src,
                                                __bf16* __restrict__ dst, int n) {
  int i = blockIdx.x * 256 + threadIdx.x;
  if (i < n) dst[i] = f2b(src[i]);
}

// ---------- LayerNorm (one block per row), fp32 in -> bf16 out ----------
__global__ __launch_bounds__(256) void ln_kernel(const float* __restrict__ x,
                                                 const float* __restrict__ w0, const float* __restrict__ b0,
                                                 const float* __restrict__ w1, const float* __restrict__ b1,
                                                 int split, __bf16* __restrict__ out) {
  const int row = blockIdx.x;
  const int n   = row & (N_ - 1);
  const float* w  = (n < split) ? w0 : w1;
  const float* bb = (n < split) ? b0 : b1;
  const float4 v = ((const float4*)(x + (size_t)row * C_))[threadIdx.x];
  float s  = v.x + v.y + v.z + v.w;
  float s2 = v.x*v.x + v.y*v.y + v.z*v.z + v.w*v.w;
#pragma unroll
  for (int off = 16; off >= 1; off >>= 1) {
    s  += __shfl_xor(s,  off, 32);
    s2 += __shfl_xor(s2, off, 32);
  }
  __shared__ float ps[8], ps2[8];
  if ((threadIdx.x & 31) == 0) { ps[threadIdx.x >> 5] = s; ps2[threadIdx.x >> 5] = s2; }
  __syncthreads();
  float ts = 0.f, ts2 = 0.f;
#pragma unroll
  for (int j = 0; j < 8; ++j) { ts += ps[j]; ts2 += ps2[j]; }
  const float mean = ts * (1.f / C_);
  const float var  = ts2 * (1.f / C_) - mean * mean;
  const float rstd = rsqrtf(var + EPS_);
  const int c0 = threadIdx.x * 4;
  __bf16* o = out + (size_t)row * C_ + c0;
  o[0] = f2b((v.x - mean) * rstd * w[c0 + 0] + bb[c0 + 0]);
  o[1] = f2b((v.y - mean) * rstd * w[c0 + 1] + bb[c0 + 1]);
  o[2] = f2b((v.z - mean) * rstd * w[c0 + 2] + bb[c0 + 2]);
  o[3] = f2b((v.w - mean) * rstd * w[c0 + 3] + bb[c0 + 3]);
}

// ---------- generic bf16 WMMA GEMM: out = A(MxK) @ Bw(NcolsxK)^T ----------
struct GemmArgs {
  const __bf16* A;  int lda;            // row index into A is (b*N + n)
  const __bf16* Bw;                     // [Ncols][K], row-major
  int K, Ncols, rows, seg, epi;         // epi: 0=qkv scatter, 1=residual f32, 2=fc1 gelu bf16
  const float* bias;
  const float* qb; const float* vb;
  const float* resid_in;
  const float* gamma;
  float* out_f32;
  __bf16* out_bf;
  __bf16* qp; __bf16* kp; __bf16* vp;   // qkv outputs [B,H,N,HD]
};

#define TM 128
#define TN 128
#define TK 32

// 256 threads = 8 waves (4x2). Each wave: 32 rows x 64 cols = 8 WMMA tiles.
// Double-buffered LDS, async global->LDS staging overlapped with WMMA.
__global__ __launch_bounds__(256) void gemm_bf16(GemmArgs ga) {
  __shared__ alignas(16) __bf16 As[2][TM * TK];   // [row][k]
  __shared__ alignas(16) __bf16 Bs[2][TN * TK];   // [n][k]
  const int tid = threadIdx.x;
  const int w  = tid >> 5, l = tid & 31;
  const int wm = w >> 1,  wn = w & 1;
  const int half = l >> 4, ln = l & 15;
  const int m0 = blockIdx.y * TM;
  const int n0 = blockIdx.x * TN;

  f32x8 acc[2][4];
#pragma unroll
  for (int mi = 0; mi < 2; ++mi)
#pragma unroll
    for (int nt = 0; nt < 4; ++nt) acc[mi][nt] = zero8();

  // Staging coordinates: 2 A chunks + 2 B chunks of 16B per thread per K-step.
  const __bf16* aptr[2];
  const __bf16* bptr[2];
  int ldsoff[2];
#pragma unroll
  for (int j = 0; j < 2; ++j) {
    const int idx = tid + j * 256;           // 0..511
    const int row = idx >> 2, ch = idx & 3;  // 128 rows x 4 chunks
    bool v_;
    const int gr = map_row(m0 + row, ga.seg, v_, ga.rows);
    aptr[j] = ga.A + (size_t)gr * ga.lda + ch * 8;
    bptr[j] = ga.Bw + (size_t)(n0 + row) * ga.K + ch * 8;
    ldsoff[j] = row * TK + ch * 8;
  }

  auto issue_tile = [&](int kk, int s) {
#pragma unroll
    for (int j = 0; j < 2; ++j) copy16_async(aptr[j] + kk, &As[s][ldsoff[j]]);
#pragma unroll
    for (int j = 0; j < 2; ++j) copy16_async(bptr[j] + kk, &Bs[s][ldsoff[j]]);
  };

  const int steps = ga.K / TK;
  issue_tile(0, 0);
  for (int i = 0; i < steps; ++i) {
    const int s = i & 1;
    if (i + 1 < steps) { issue_tile((i + 1) * TK, s ^ 1); async_wait<4>(); }
    else               { async_wait<0>(); }
    __syncthreads();

    bf16x16 af[2];
#pragma unroll
    for (int mi = 0; mi < 2; ++mi) {
      const int arow = wm * 32 + mi * 16 + ln;
      ((uint4*)&af[mi])[0] = *(uint4*)&As[s][arow * TK + half * 8];
      ((uint4*)&af[mi])[1] = *(uint4*)&As[s][arow * TK + 16 + half * 8];
    }
#pragma unroll
    for (int nt = 0; nt < 4; ++nt) {
      bf16x16 bfg;
      const int bcol = wn * 64 + nt * 16 + ln;
      ((uint4*)&bfg)[0] = *(uint4*)&Bs[s][bcol * TK + half * 8];
      ((uint4*)&bfg)[1] = *(uint4*)&Bs[s][bcol * TK + 16 + half * 8];
#pragma unroll
      for (int mi = 0; mi < 2; ++mi)
        acc[mi][nt] = __builtin_amdgcn_wmma_f32_16x16x32_bf16(false, af[mi], false, bfg,
                                                              (short)0, acc[mi][nt], false, false);
    }
    __syncthreads();
  }

  // Epilogue. C/D layout: VGPR i -> row = half*8 + i, col = l%16.
#pragma unroll
  for (int mi = 0; mi < 2; ++mi) {
#pragma unroll
    for (int nt = 0; nt < 4; ++nt) {
      const int col = n0 + wn * 64 + nt * 16 + ln;
#pragma unroll
      for (int i = 0; i < 8; ++i) {
        const int rloc = m0 + wm * 32 + mi * 16 + half * 8 + i;
        bool valid; const int gr = map_row(rloc, ga.seg, valid, ga.rows);
        if (!valid) continue;
        const float vaccum = acc[mi][nt][i];
        if (ga.epi == 0) {                       // QKV scatter
          const int which = col >> 10, cc = col & 1023;
          float bias = (which == 0) ? ga.qb[cc] : ((which == 2) ? ga.vb[cc] : 0.f);
          float val = vaccum + bias;
          if (which == 0) val *= SCALE_;
          const int b = gr >> 11, n = gr & (N_ - 1);
          const int h = cc >> 6, d = cc & 63;
          __bf16* dst = (which == 0) ? ga.qp : ((which == 1) ? ga.kp : ga.vp);
          dst[(((size_t)b * H_ + h) * N_ + n) * HD_ + d] = f2b(val);
        } else if (ga.epi == 1) {                // out = resid + gamma * (acc + bias)
          const float val = vaccum + ga.bias[col];
          const size_t off = (size_t)gr * C_ + col;
          ga.out_f32[off] = ga.resid_in[off] + ga.gamma[col] * val;
        } else {                                 // fc1: GELU(acc + bias) -> bf16
          const float val = vaccum + ga.bias[col];
          const float g = 0.5f * val * (1.f + erff(val * 0.70710678118654752f));
          ga.out_bf[(size_t)gr * HID_ + col] = f2b(g);
        }
      }
    }
  }
}

// ---------- flash attention: softmax(q k^T + bias, mask) @ v ----------
// grid = (N/64, B*H), block = 128 (4 waves). Wave w owns 16 query rows.
// Q and K stream via async LDS copies (K double-buffered); V goes through
// VGPRs because it must be transposed, overlapping the async K traffic.
__global__ __launch_bounds__(128) void attn_kernel(const __bf16* __restrict__ q,
                                                   const __bf16* __restrict__ k,
                                                   const __bf16* __restrict__ v,
                                                   const float* __restrict__ rpb,
                                                   const int* __restrict__ mask,
                                                   __bf16* __restrict__ out) {
  const int bh = blockIdx.y;
  const int b = bh >> 4, h = bh & 15;
  const int q0 = blockIdx.x * 64;
  const int tid = threadIdx.x;
  const int w = tid >> 5, l = tid & 31, half = l >> 4, ln = l & 15;

  __shared__ alignas(16) __bf16 Qs[64 * 64];
  __shared__ alignas(16) __bf16 Ks[2][64 * 64];
  __shared__ alignas(16) __bf16 Vts[64 * 64];        // transposed: [d][key]
  __shared__ alignas(16) __bf16 Ps[4][16 * 64];      // per-wave probabilities

  const __bf16* qp = q + ((size_t)bh * N_ + q0) * HD_;
#pragma unroll
  for (int r = 0; r < 4; ++r) {
    const int idx = tid + r * 128;
    copy16_async(qp + idx * 8, &Qs[idx * 8]);
  }
  auto issue_k = [&](int kb, int s) {
    const __bf16* kp = k + ((size_t)bh * N_ + kb) * HD_;
#pragma unroll
    for (int r = 0; r < 4; ++r) {
      const int idx = tid + r * 128;
      copy16_async(kp + idx * 8, &Ks[s][idx * 8]);
    }
  };
  issue_k(0, 0);

  f32x8 o[4];
#pragma unroll
  for (int i = 0; i < 4; ++i) o[i] = zero8();
  float mprev[8], lsum[8];
#pragma unroll
  for (int i = 0; i < 8; ++i) { mprev[i] = -INFINITY; lsum[i] = 0.f; }

  for (int kb = 0; kb < N_; kb += 64) {
    const int s = (kb >> 6) & 1;
    const bool more = (kb + 64 < N_);
    if (more) issue_k(kb + 64, s ^ 1);

    // V tile: load + transpose through VGPRs (overlaps async K DMA)
    const __bf16* vp = v + ((size_t)bh * N_ + kb) * HD_;
#pragma unroll
    for (int r = 0; r < 4; ++r) {
      const int idx = tid + r * 128;
      const int key = idx >> 3, ch = idx & 7;
      uint4 t = ((const uint4*)(vp + (size_t)key * HD_))[ch];
      const __bf16* tb = (const __bf16*)&t;
#pragma unroll
      for (int j = 0; j < 8; ++j) Vts[(ch * 8 + j) * 64 + key] = tb[j];
    }
    if (more) async_wait<4>(); else async_wait<0>();
    __syncthreads();

    // S = Q @ K^T  (16 rows x 64 keys per wave)
    f32x8 sreg[4];
#pragma unroll
    for (int nt = 0; nt < 4; ++nt) {
      f32x8 a = zero8();
#pragma unroll
      for (int kc = 0; kc < 64; kc += 32) {
        bf16x16 af, bfg;
        ((uint4*)&af)[0]  = *(uint4*)&Qs[(w * 16 + ln) * 64 + kc + half * 8];
        ((uint4*)&af)[1]  = *(uint4*)&Qs[(w * 16 + ln) * 64 + kc + 16 + half * 8];
        ((uint4*)&bfg)[0] = *(uint4*)&Ks[s][(nt * 16 + ln) * 64 + kc + half * 8];
        ((uint4*)&bfg)[1] = *(uint4*)&Ks[s][(nt * 16 + ln) * 64 + kc + 16 + half * 8];
        a = __builtin_amdgcn_wmma_f32_16x16x32_bf16(false, af, false, bfg, (short)0, a, false, false);
      }
      sreg[nt] = a;
    }

    // bias + mask
#pragma unroll
    for (int nt = 0; nt < 4; ++nt) {
      const int col = kb + nt * 16 + ln;
      const int mk = mask[b * N_ + col];
#pragma unroll
      for (int i = 0; i < 8; ++i) {
        const int row = q0 + w * 16 + half * 8 + i;
        float sv = sreg[nt][i] + rpb[((size_t)h * N_ + row) * N_ + col];
        sreg[nt][i] = mk ? sv : -INFINITY;
      }
    }

    // Online softmax (row reduction across the 16 lanes sharing a row)
    float alpha[8];
#pragma unroll
    for (int i = 0; i < 8; ++i) {
      float mx = fmaxf(fmaxf(sreg[0][i], sreg[1][i]), fmaxf(sreg[2][i], sreg[3][i]));
#pragma unroll
      for (int off = 1; off < 16; off <<= 1) mx = fmaxf(mx, __shfl_xor(mx, off, 32));
      const float mnew = fmaxf(mprev[i], mx);
      alpha[i] = expf(mprev[i] - mnew);
      float rs = 0.f;
#pragma unroll
      for (int nt = 0; nt < 4; ++nt) {
        const float p = expf(sreg[nt][i] - mnew);
        sreg[nt][i] = p; rs += p;
      }
#pragma unroll
      for (int off = 1; off < 16; off <<= 1) rs += __shfl_xor(rs, off, 32);
      lsum[i] = lsum[i] * alpha[i] + rs;
      mprev[i] = mnew;
    }

    // Rescale O, stage P to LDS (C-layout -> A-layout round trip)
#pragma unroll
    for (int dt = 0; dt < 4; ++dt)
#pragma unroll
      for (int i = 0; i < 8; ++i) o[dt][i] = o[dt][i] * alpha[i];
#pragma unroll
    for (int nt = 0; nt < 4; ++nt)
#pragma unroll
      for (int i = 0; i < 8; ++i)
        Ps[w][(half * 8 + i) * 64 + nt * 16 + ln] = f2b(sreg[nt][i]);

    // O += P @ V
#pragma unroll
    for (int dt = 0; dt < 4; ++dt) {
#pragma unroll
      for (int kc = 0; kc < 64; kc += 32) {
        bf16x16 af, bfg;
        ((uint4*)&af)[0]  = *(uint4*)&Ps[w][ln * 64 + kc + half * 8];
        ((uint4*)&af)[1]  = *(uint4*)&Ps[w][ln * 64 + kc + 16 + half * 8];
        ((uint4*)&bfg)[0] = *(uint4*)&Vts[(dt * 16 + ln) * 64 + kc + half * 8];
        ((uint4*)&bfg)[1] = *(uint4*)&Vts[(dt * 16 + ln) * 64 + kc + 16 + half * 8];
        o[dt] = __builtin_amdgcn_wmma_f32_16x16x32_bf16(false, af, false, bfg, (short)0, o[dt], false, false);
      }
    }
    __syncthreads();
  }

  // Normalize and write [B, N, C] bf16
#pragma unroll
  for (int dt = 0; dt < 4; ++dt)
#pragma unroll
    for (int i = 0; i < 8; ++i) {
      const int row = q0 + w * 16 + half * 8 + i;
      const int d = dt * 16 + ln;
      out[((size_t)(b * N_ + row)) * C_ + h * HD_ + d] = f2b(o[dt][i] / lsum[i]);
    }
}

// ---------- host orchestration ----------
extern "C" void kernel_launch(void* const* d_in, const int* in_sizes, int n_in,
                              void* d_out, int out_size, void* d_ws, size_t ws_size,
                              hipStream_t stream) {
  (void)in_sizes; (void)n_in; (void)out_size; (void)ws_size;
  const float* x     = (const float*)d_in[0];
  const int*   mask  = (const int*)d_in[1];
  const float* rpb   = (const float*)d_in[2];
  const float* n1w   = (const float*)d_in[3];
  const float* n1b   = (const float*)d_in[4];
  const float* qkvw  = (const float*)d_in[5];
  const float* qbias = (const float*)d_in[6];
  const float* vbias = (const float*)d_in[7];
  const float* projw = (const float*)d_in[8];
  const float* projb = (const float*)d_in[9];
  const float* g1    = (const float*)d_in[10];
  const float* g2    = (const float*)d_in[11];
  const float* n2tw  = (const float*)d_in[12];
  const float* n2tb  = (const float*)d_in[13];
  const float* t1w   = (const float*)d_in[14];
  const float* t1b   = (const float*)d_in[15];
  const float* t2w   = (const float*)d_in[16];
  const float* t2b   = (const float*)d_in[17];
  const float* n2iw  = (const float*)d_in[18];
  const float* n2ib  = (const float*)d_in[19];
  const float* i1w   = (const float*)d_in[20];
  const float* i1b   = (const float*)d_in[21];
  const float* i2w   = (const float*)d_in[22];
  const float* i2b   = (const float*)d_in[23];
  float* outp = (float*)d_out;

  const size_t BNC = (size_t)B_ * N_ * C_;
  char* p = (char*)d_ws;
  auto carve = [&](size_t bytes) { char* r = p; p += (bytes + 255) & ~(size_t)255; return (void*)r; };
  __bf16* wqkv  = (__bf16*)carve((size_t)3 * C_ * C_ * 2);
  __bf16* wproj = (__bf16*)carve((size_t)C_ * C_ * 2);
  __bf16* wt1   = (__bf16*)carve((size_t)HID_ * C_ * 2);
  __bf16* wt2   = (__bf16*)carve((size_t)C_ * HID_ * 2);
  __bf16* wi1   = (__bf16*)carve((size_t)HID_ * C_ * 2);
  __bf16* wi2   = (__bf16*)carve((size_t)C_ * HID_ * 2);
  __bf16* hbf   = (__bf16*)carve(BNC * 2);
  __bf16* qbuf  = (__bf16*)carve(BNC * 2);
  __bf16* kbuf  = (__bf16*)carve(BNC * 2);
  __bf16* vbuf  = (__bf16*)carve(BNC * 2);
  __bf16* attno = (__bf16*)carve(BNC * 2);
  float*  x1    = (float*) carve(BNC * 4);
  __bf16* h2    = (__bf16*)carve(BNC * 2);
  __bf16* h3    = (__bf16*)carve((size_t)B_ * N_ * HID_ * 2);

  auto cvt = [&](const float* s, __bf16* d, int n) {
    cvt_bf16<<<(n + 255) / 256, 256, 0, stream>>>(s, d, n);
  };
  cvt(qkvw,  wqkv,  3 * C_ * C_);
  cvt(projw, wproj, C_ * C_);
  cvt(t1w, wt1, HID_ * C_);
  cvt(t2w, wt2, C_ * HID_);
  cvt(i1w, wi1, HID_ * C_);
  cvt(i2w, wi2, C_ * HID_);

  // LN1 -> bf16
  ln_kernel<<<B_ * N_, 256, 0, stream>>>(x, n1w, n1b, n1w, n1b, N_, hbf);

  // QKV GEMM
  {
    GemmArgs g = {};
    g.A = hbf; g.lda = C_; g.Bw = wqkv; g.K = C_; g.Ncols = 3 * C_;
    g.rows = B_ * N_; g.seg = 0; g.epi = 0;
    g.qb = qbias; g.vb = vbias; g.qp = qbuf; g.kp = kbuf; g.vp = vbuf;
    gemm_bf16<<<dim3((3 * C_) / TN, (B_ * N_) / TM), 256, 0, stream>>>(g);
  }

  // Attention
  attn_kernel<<<dim3(N_ / 64, B_ * H_), 128, 0, stream>>>(qbuf, kbuf, vbuf, rpb, mask, attno);

  // Proj GEMM + layerscale residual -> x1
  {
    GemmArgs g = {};
    g.A = attno; g.lda = C_; g.Bw = wproj; g.K = C_; g.Ncols = C_;
    g.rows = B_ * N_; g.seg = 0; g.epi = 1;
    g.bias = projb; g.resid_in = x; g.gamma = g1; g.out_f32 = x1;
    gemm_bf16<<<dim3(C_ / TN, (B_ * N_) / TM), 256, 0, stream>>>(g);
  }

  // LN2 (token params if n < MT, else image params)
  ln_kernel<<<B_ * N_, 256, 0, stream>>>(x1, n2tw, n2tb, n2iw, n2ib, MT_, h2);

  const int trows = B_ * MT_;          // 160
  const int irows = B_ * (N_ - MT_);   // 8032
  auto mt = [](int r) { return (r + TM - 1) / TM; };

  // FC1 (GELU) per segment -> h3
  {
    GemmArgs g = {};
    g.A = h2; g.lda = C_; g.Bw = wt1; g.K = C_; g.Ncols = HID_;
    g.rows = trows; g.seg = 1; g.epi = 2; g.bias = t1b; g.out_bf = h3;
    gemm_bf16<<<dim3(HID_ / TN, mt(trows)), 256, 0, stream>>>(g);
    g.Bw = wi1; g.rows = irows; g.seg = 2; g.bias = i1b;
    gemm_bf16<<<dim3(HID_ / TN, mt(irows)), 256, 0, stream>>>(g);
  }

  // FC2 + layerscale residual per segment -> d_out
  {
    GemmArgs g = {};
    g.A = h3; g.lda = HID_; g.Bw = wt2; g.K = HID_; g.Ncols = C_;
    g.rows = trows; g.seg = 1; g.epi = 1;
    g.bias = t2b; g.resid_in = x1; g.gamma = g2; g.out_f32 = outp;
    gemm_bf16<<<dim3(C_ / TN, mt(trows)), 256, 0, stream>>>(g);
    g.Bw = wi2; g.rows = irows; g.seg = 2; g.bias = i2b;
    gemm_bf16<<<dim3(C_ / TN, mt(irows)), 256, 0, stream>>>(g);
  }
}
